// Transformer_23055384445649
// MI455X (gfx1250) — compile-verified, matrix operand order checked by reading, not run on hardware
//
#include <hip/hip_runtime.h>
#include <hip/hip_bf16.h>

typedef __attribute__((ext_vector_type(16))) _Float16     v16h;
typedef __attribute__((ext_vector_type(8)))  float        v8f;
typedef __attribute__((ext_vector_type(8)))  unsigned int v8u;

#define LQ 300
#define SK 1024
#define NBATCH 32
#define EMB 256
#define HEADS 8
#define DH 32

// Load a 16x32 f16 A/B fragment from a tile whose row data (32 halves for the
// relevant K-group) starts at rowp. grp = lane>>4 selects K-half per ISA layout:
//   j<4 : K = 2j   + 8*grp    -> u32 index j + 4*grp
//   j>=4: K = 16+2(j-4)+8*grp -> u32 index 8 + (j-4) + 4*grp
__device__ __forceinline__ v16h load_frag(const unsigned int* __restrict__ rowp, int grp) {
    v8u t;
    int o = 4 * grp;
    t[0] = rowp[o + 0];  t[1] = rowp[o + 1];
    t[2] = rowp[o + 2];  t[3] = rowp[o + 3];
    t[4] = rowp[o + 8];  t[5] = rowp[o + 9];
    t[6] = rowp[o + 10]; t[7] = rowp[o + 11];
    return __builtin_bit_cast(v16h, t);
}

// Async global -> LDS copy of 16 bytes (per lane), tracked by ASYNCcnt.
// lds: workgroup-relative LDS byte address (low 32 bits of the flat pointer,
// which per the ISA flat->LDS mapping is the LDS offset).
__device__ __forceinline__ void async_copy16(unsigned lds, const void* gaddr) {
    asm volatile("global_load_async_to_lds_b128 %0, %1, off"
                 :: "v"(lds), "v"((unsigned long long)(uintptr_t)gaddr)
                 : "memory");
}

// ---------------------------------------------------------------------------
// Generic WMMA GEMM:  Y[M][256] = X[M][256] @ W^T + bias
// Block: 256 threads (8 waves). Tile: 64 (M) x 128 (N). K stepped by 32.
// mode 0: out f32 row-major [M][256]                (final projection)
// mode 1: out f16 Qh  [(nb*8+h)*rows + img][32], scaled
// mode 2: out f16 Kh  [(nb*8+h)*rows + img][32]
// mode 3: out f16 Vt  [(nb*8+h)*32 + d][1024]  (transposed over S)
// where m = img*32 + nb, n = h*32 + d.
// ---------------------------------------------------------------------------
__global__ __launch_bounds__(256) void gemm_kernel(
    const float* __restrict__ X, const float* __restrict__ W,
    const float* __restrict__ bias, void* __restrict__ out,
    int M, int mode, float scale, int rows)
{
    __shared__ alignas(16) _Float16 Xs[64 * 32];
    __shared__ alignas(16) _Float16 Ws[128 * 32];

    const int m0 = blockIdx.x * 64;
    const int n0 = blockIdx.y * 128;
    const int t = threadIdx.x;
    const int w = t >> 5, lane = t & 31;
    const int mi = w & 3, ni = w >> 2;          // wave -> 16-row, 64-col subtile
    const int grp = lane >> 4, nl = lane & 15;

    v8f acc[4];
#pragma unroll
    for (int j = 0; j < 4; ++j) acc[j] = (v8f){0.f,0.f,0.f,0.f,0.f,0.f,0.f,0.f};

    for (int ks = 0; ks < EMB; ks += 32) {
        __syncthreads();
        // stage X tile 64x32 (f32 -> f16)
#pragma unroll
        for (int i = 0; i < 2; ++i) {
            int idx = t + i * 256;              // 0..511
            int row = idx >> 3, c4 = idx & 7;
            float4 xv = *(const float4*)(X + (size_t)(m0 + row) * EMB + ks + c4 * 4);
            _Float16* dst = Xs + row * 32 + c4 * 4;
            dst[0] = (_Float16)xv.x; dst[1] = (_Float16)xv.y;
            dst[2] = (_Float16)xv.z; dst[3] = (_Float16)xv.w;
        }
        // stage W tile 128x32 (f32 -> f16)
#pragma unroll
        for (int i = 0; i < 4; ++i) {
            int idx = t + i * 256;              // 0..1023
            int row = idx >> 3, c4 = idx & 7;
            float4 wv = *(const float4*)(W + (size_t)(n0 + row) * EMB + ks + c4 * 4);
            _Float16* dst = Ws + row * 32 + c4 * 4;
            dst[0] = (_Float16)wv.x; dst[1] = (_Float16)wv.y;
            dst[2] = (_Float16)wv.z; dst[3] = (_Float16)wv.w;
        }
        __syncthreads();

        const unsigned int* xb = (const unsigned int*)Xs;
        const unsigned int* wb = (const unsigned int*)Ws;
        v16h a = load_frag(xb + (mi * 16 + nl) * 16, grp);
#pragma unroll
        for (int j = 0; j < 4; ++j) {
            v16h b = load_frag(wb + (ni * 64 + j * 16 + nl) * 16, grp);
            acc[j] = __builtin_amdgcn_wmma_f32_16x16x32_f16(
                false, a, false, b, (short)0, acc[j], false, false);
        }
    }

    // epilogue
#pragma unroll
    for (int j = 0; j < 4; ++j) {
#pragma unroll
        for (int r = 0; r < 8; ++r) {
            int m = m0 + mi * 16 + grp * 8 + r;
            int nn = n0 + ni * 64 + j * 16 + nl;
            if (m >= M) continue;
            float val = acc[j][r] + bias[nn];
            if (mode == 0) {
                ((float*)out)[(size_t)m * EMB + nn] = val;
            } else {
                int nb = m & 31, img = m >> 5;
                int h = nn >> 5, d = nn & 31;
                int bb = nb * HEADS + h;
                _Float16 hv = (_Float16)(val * scale);
                if (mode == 3)
                    ((_Float16*)out)[((size_t)bb * DH + d) * SK + img] = hv;
                else
                    ((_Float16*)out)[((size_t)bb * rows + img) * DH + d] = hv;
            }
        }
    }
}

// ---------------------------------------------------------------------------
// Fused flash attention: per block = one head b; 4 waves, each owns one
// 16-row L tile. S looped in 64-wide chunks, staged to LDS with async
// global->LDS b128 copies (ASYNCcnt). gaussian streamed once with NT hints.
// ---------------------------------------------------------------------------
__global__ __launch_bounds__(128) void attn_kernel(
    const _Float16* __restrict__ Qh, const _Float16* __restrict__ Kh,
    const _Float16* __restrict__ Vt, const float* __restrict__ gauss,
    const unsigned char* __restrict__ mask, float* __restrict__ Of)
{
    __shared__ alignas(16) _Float16 Ks[64 * 32];     // [s_local][dh]
    __shared__ alignas(16) _Float16 Vs[32 * 64];     // [d][s_local]
    __shared__ alignas(16) _Float16 Ps[4][16 * 64];  // per-wave P tile [m][64]

    const int b = blockIdx.x;                 // head-batch index (0..255)
    const int nb = b >> 3, h = b & 7;
    const int w = threadIdx.x >> 5, lane = threadIdx.x & 31;
    const int tile = blockIdx.y * 4 + w;      // L tile (0..19, 19 invalid)
    const int l0 = tile * 16;
    const int grp = lane >> 4, n = lane & 15;

    // Q fragment: A layout row = lane&15 (clamped for partial/overflow tiles)
    const int lrow = min(l0 + n, LQ - 1);
    const unsigned int* qb = (const unsigned int*)(Qh + (size_t)b * LQ * DH);
    v16h aq = load_frag(qb + lrow * 16, grp);

    const float* gl = gauss + (size_t)b * LQ * SK;
    const unsigned char* mrowbase = mask + (size_t)nb * SK;

    float mmax[8], lsum[8];
#pragma unroll
    for (int r = 0; r < 8; ++r) { mmax[r] = -1e30f; lsum[r] = 0.f; }
    v8f o0 = (v8f){0.f,0.f,0.f,0.f,0.f,0.f,0.f,0.f};
    v8f o1 = (v8f){0.f,0.f,0.f,0.f,0.f,0.f,0.f,0.f};

    for (int sc = 0; sc < SK / 64; ++sc) {
        const int s0 = sc * 64;
        __syncthreads();
        // --- async stage: K chunk [64 s][32 dh] and V^T chunk [32 d][64 s] ---
#pragma unroll
        for (int i = 0; i < 2; ++i) {
            int slot = threadIdx.x + 128 * i;   // 0..255
            int krow = slot >> 2, kpart = slot & 3;
            async_copy16((unsigned)(uintptr_t)(Ks + krow * 32 + kpart * 8),
                         Kh + ((size_t)b * SK + s0 + krow) * DH + kpart * 8);
            int vrow = slot >> 3, vpart = slot & 7;
            async_copy16((unsigned)(uintptr_t)(Vs + vrow * 64 + vpart * 8),
                         Vt + ((size_t)b * DH + vrow) * SK + s0 + vpart * 8);
        }
        asm volatile("s_wait_asynccnt 0" ::: "memory");
        __syncthreads();

        // --- scores: Q(16x32) @ K^T, 4 WMMA tiles covering s0..s0+63 ---
        const unsigned int* kb = (const unsigned int*)Ks;
        v8f z = (v8f){0.f,0.f,0.f,0.f,0.f,0.f,0.f,0.f};
        v8f c[4];
#pragma unroll
        for (int j = 0; j < 4; ++j) {
            v16h bk = load_frag(kb + (j * 16 + n) * 16, grp);
            c[j] = __builtin_amdgcn_wmma_f32_16x16x32_f16(
                false, aq, false, bk, (short)0, z, false, false);
        }

        float mk[4];
#pragma unroll
        for (int j = 0; j < 4; ++j)
            mk[j] = mrowbase[s0 + j * 16 + n] ? -1e30f : 0.f;

        // --- gaussian bias (NT stream) + online softmax ---
#pragma unroll
        for (int r = 0; r < 8; ++r) {
            int l = min(l0 + grp * 8 + r, LQ - 1);
            const float* gr = gl + (size_t)l * SK + s0;
            float v[4];
#pragma unroll
            for (int j = 0; j < 4; ++j)
                v[j] = c[j][r] + __builtin_nontemporal_load(gr + j * 16 + n) + mk[j];

            float rmax = fmaxf(fmaxf(v[0], v[1]), fmaxf(v[2], v[3]));
#pragma unroll
            for (int xm = 1; xm < 16; xm <<= 1) rmax = fmaxf(rmax, __shfl_xor(rmax, xm));
            float mnew  = fmaxf(mmax[r], rmax);
            float alpha = __expf(mmax[r] - mnew);
            float ps = 0.f;
#pragma unroll
            for (int j = 0; j < 4; ++j) {
                float p = __expf(v[j] - mnew);
                ps += p;
                Ps[w][(grp * 8 + r) * 64 + j * 16 + n] = (_Float16)p;
            }
#pragma unroll
            for (int xm = 1; xm < 16; xm <<= 1) ps += __shfl_xor(ps, xm);
            lsum[r] = lsum[r] * alpha + ps;
            mmax[r] = mnew;
            o0[r] *= alpha;
            o1[r] *= alpha;
        }
        // same-wave LDS ordering + compiler barrier before cross-lane reload
        asm volatile("s_wait_dscnt 0" ::: "memory");

        // --- O += P(16x64) @ V(64x32): 2 K-halves x 2 N-halves ---
        const unsigned int* pb = (const unsigned int*)Ps[w];
        const unsigned int* vb = (const unsigned int*)Vs;
#pragma unroll
        for (int kk = 0; kk < 2; ++kk) {
            v16h ap  = load_frag(pb + n * 32 + kk * 16, grp);
            v16h bv0 = load_frag(vb + n * 32 + kk * 16, grp);
            v16h bv1 = load_frag(vb + (n + 16) * 32 + kk * 16, grp);
            o0 = __builtin_amdgcn_wmma_f32_16x16x32_f16(false, ap, false, bv0, (short)0, o0, false, false);
            o1 = __builtin_amdgcn_wmma_f32_16x16x32_f16(false, ap, false, bv1, (short)0, o1, false, false);
        }
    }

    // normalize + store O as f32 [l][nb][e], e = h*32 + d
#pragma unroll
    for (int r = 0; r < 8; ++r) {
        int l = l0 + grp * 8 + r;
        if (l < LQ) {
            float inv = 1.f / lsum[r];
            size_t base = ((size_t)l * NBATCH + nb) * EMB + h * DH;
            Of[base + n]      = o0[r] * inv;
            Of[base + 16 + n] = o1[r] * inv;
        }
    }
}

extern "C" void kernel_launch(void* const* d_in, const int* in_sizes, int n_in,
                              void* d_out, int out_size, void* d_ws, size_t ws_size,
                              hipStream_t stream) {
    const float* query = (const float*)d_in[0];
    const float* key   = (const float*)d_in[1];
    const float* value = (const float*)d_in[2];
    const float* gauss = (const float*)d_in[3];
    const unsigned char* mask = (const unsigned char*)d_in[4];
    const float* inW  = (const float*)d_in[5];   // (3E, E)
    const float* inB  = (const float*)d_in[6];   // (3E,)
    const float* outW = (const float*)d_in[7];   // (E, E)
    const float* outB = (const float*)d_in[8];   // (E,)

    char* ws = (char*)d_ws;
    const size_t QH_BYTES = (size_t)256 * LQ * DH * sizeof(_Float16);   // 4.9 MB
    const size_t KH_BYTES = (size_t)256 * SK * DH * sizeof(_Float16);   // 16.8 MB
    _Float16* Qh = (_Float16*)ws;
    _Float16* Kh = (_Float16*)(ws + QH_BYTES);
    _Float16* Vt = (_Float16*)(ws + QH_BYTES + KH_BYTES);
    float*    Of = (float*)   (ws + QH_BYTES + 2 * KH_BYTES);

    const float scaling = 0.17677669529663687f;  // 32^-0.5
    dim3 blk(256);

    // Q/K/V projections (WMMA GEMMs, f16 outputs in attention-friendly layouts)
    gemm_kernel<<<dim3(150, 2), blk, 0, stream>>>(query, inW,           inB,       Qh, 9600,  1, scaling, LQ);
    gemm_kernel<<<dim3(512, 2), blk, 0, stream>>>(key,   inW + 65536,   inB + 256, Kh, 32768, 2, 1.0f,    SK);
    gemm_kernel<<<dim3(512, 2), blk, 0, stream>>>(value, inW + 131072,  inB + 512, Vt, 32768, 3, 1.0f,    SK);

    // Fused flash attention with gaussian bias (reads 314 MB gaussian once, NT)
    attn_kernel<<<dim3(256, 5), dim3(128), 0, stream>>>(Qh, Kh, Vt, gauss, mask, Of);

    // Output projection -> d_out f32 (L, N, E)
    gemm_kernel<<<dim3(150, 2), blk, 0, stream>>>(Of, outW, outB, (float*)d_out, 9600, 0, 1.0f, LQ);
}
